// SelfRetentionV2_53120155517202
// MI455X (gfx1250) — matrix operand order repaired and checked by reading, not compiled, not run on hardware
//
#include <hip/hip_runtime.h>

// ---------------------------------------------------------------------------
// SelfRetention forward chunk for MI455X (gfx1250, wave32, WMMA bf16 +
// async global->LDS staging of the decay-mask tiles)
// ---------------------------------------------------------------------------

typedef __bf16 bf16_t;
typedef __attribute__((ext_vector_type(16))) __bf16 v16bf;
typedef __attribute__((ext_vector_type(8)))  __bf16 v8bf;
typedef __attribute__((ext_vector_type(4)))  __bf16 v4bf;
typedef __attribute__((ext_vector_type(8)))  float  v8f;

#define WMMA_BF16(A, Bm, Cm) \
  __builtin_amdgcn_wmma_f32_16x16x32_bf16(false, (A), false, (Bm), (short)0, (Cm), false, false)

constexpr int B_  = 4;
constexpr int H_  = 8;
constexpr int C_  = 2048;
constexpr int D1_ = 256;
constexpr int D2_ = 512;
constexpr int BM  = 64;   // query rows per workgroup
constexpr int BN  = 64;   // key/value rows per inner step

// fp32 -> bf16, round-to-nearest-even (pure integer path)
__device__ __forceinline__ bf16_t f2bf(float x) {
  unsigned u = __builtin_bit_cast(unsigned, x);
  unsigned r = (u + 0x7FFFu + ((u >> 16) & 1u)) >> 16;
  unsigned short s = (unsigned short)r;
  return __builtin_bit_cast(bf16_t, s);
}

// Async 16B global -> LDS copy (per lane), tracked by ASYNCcnt.
__device__ __forceinline__ void async_g2l_b128(void* lds_ptr, const void* gaddr) {
  unsigned lds_off = (unsigned)(unsigned long long)lds_ptr;  // low 32 bits = LDS offset
  asm volatile("global_load_async_to_lds_b128 %0, %1, off"
               :: "v"(lds_off), "v"(gaddr) : "memory");
}
__device__ __forceinline__ void wait_async0() {
  asm volatile("s_wait_asynccnt 0x0" ::: "memory");
}

// A-fragment (16x32 bf16) per ISA 7.12.2: lanes 0-15 -> M=lane, K pairs 0..7 in v0..3,
// K 16..23 in v4..7; lanes 16-31 same M, K base +8.
__device__ __forceinline__ v16bf frag_a(const bf16_t* lds, int row0, int ld, int kbase, int lane) {
  const int m  = lane & 15;
  const int kb = kbase + ((lane >> 4) << 3);
  const bf16_t* p = lds + (row0 + m) * ld + kb;
  v8bf lo = *(const v8bf*)(p);
  v8bf hi = *(const v8bf*)(p + 16);
  v16bf f;
#pragma unroll
  for (int t = 0; t < 8; ++t) { f[t] = lo[t]; f[t + 8] = hi[t]; }
  return f;
}

// B-fragment (32x16 bf16): lane l holds column N=l%16, K in table order (8h+t, 16+8h+t).
__device__ __forceinline__ v16bf frag_b(const bf16_t* lds, int krow0, int ld, int n0, int lane) {
  const int kb = krow0 + ((lane >> 4) << 3);
  const bf16_t* p = lds + kb * ld + n0 + (lane & 15);
  v16bf f;
#pragma unroll
  for (int t = 0; t < 8; ++t) { f[t] = p[t * ld]; f[t + 8] = p[(t + 16) * ld]; }
  return f;
}

// ---------------------------------------------------------------------------
// Kernel 1: o = RMSNorm( ((qn kT) .* mask) v / max(|rowsum|,1) ), transposed out
// grid: (C/BM, B*H), block: 256 threads = 8 waves
// ---------------------------------------------------------------------------
__global__ __launch_bounds__(256)
void retention_main(const float* __restrict__ q, const float* __restrict__ k,
                    const float* __restrict__ v, const float* __restrict__ mask,
                    const float* __restrict__ norml, float* __restrict__ out) {
  __shared__ bf16_t sQ [BM * D1_];   // 32 KB   qn, row-major
  __shared__ bf16_t sKT[D1_ * BN];   // 32 KB   k transposed: [a][j]
  __shared__ bf16_t sV [BN * D2_];   // 64 KB   v, row-major
  __shared__ bf16_t sS [BM * BN];    //  8 KB   masked scores (bf16)
  __shared__ float  sM [BM * BN];    // 16 KB   decay-mask tile (async-loaded f32)
  __shared__ float  sRed[BM];        // row reductions

  const int tid  = threadIdx.x;
  const int lane = tid & 31;
  const int wave = tid >> 5;
  const int rt   = wave >> 1;    // row tile 0..3  (16 rows each)
  const int ch   = wave & 1;     // column half
  const int bh   = blockIdx.y;
  const int b    = bh >> 3;
  const int h    = bh & 7;
  const int i0   = blockIdx.x * BM;

  const float* qg = q + ((size_t)bh * C_ + i0) * D1_;
  const float* kg = k + (size_t)bh * C_ * D1_;
  const float* vg = v + (size_t)bh * C_ * D2_;
  const float* nr = norml + (size_t)h * C_ + i0;
  const float* mh = mask + (size_t)h * C_ * C_;

  // ---- stage qn = q / mask_normlizer as bf16 (block is contiguous in memory)
  for (int e = tid * 4; e < BM * D1_; e += 1024) {
    const int r = e >> 8;                       // /D1_
    const float4 qv = *(const float4*)(qg + e);
    const float inv = 1.0f / nr[r];
    v4bf w;
    w[0] = f2bf(qv.x * inv); w[1] = f2bf(qv.y * inv);
    w[2] = f2bf(qv.z * inv); w[3] = f2bf(qv.w * inv);
    *(v4bf*)(sQ + e) = w;
  }

  v8f accN[16];
#pragma unroll
  for (int t = 0; t < 16; ++t) accN[t] = (v8f){0.f,0.f,0.f,0.f,0.f,0.f,0.f,0.f};
  float dsum[8] = {0.f,0.f,0.f,0.f,0.f,0.f,0.f,0.f};

  const int mrow0 = rt * 16 + ((lane >> 4) << 3);  // accumulator row base for this lane

  // ---- causal loop over key/value blocks
  for (int jb = 0; jb <= i0; jb += BN) {
    __syncthreads();   // protect sKT/sV/sS/sM from previous iteration's readers

    // ---- kick off async copy of the 64x64 f32 mask tile; it completes under GEMM1
#pragma unroll
    for (int rnd = 0; rnd < 4; ++rnd) {
      const int g   = (tid + rnd * 256) * 4;   // element index in 64x64 tile
      const int row = g >> 6;
      const int col = g & 63;
      async_g2l_b128(sM + g, mh + (size_t)(i0 + row) * C_ + jb + col);
    }

    if (jb + BN <= i0) {  // speculative prefetch of next k/v block -> global_prefetch_b8
      __builtin_prefetch(kg + (size_t)(jb + BN) * D1_ + tid * 16, 0, 1);
      __builtin_prefetch(vg + (size_t)(jb + BN) * D2_ + tid * 32, 0, 1);
    }

    // ---- stage k^T (bf16) in 4x4 micro-tiles: 4x b128 loads -> 4x b64 stores
    for (int mt = tid; mt < (BN / 4) * (D1_ / 4); mt += 256) {
      const int a0 = (mt & 63) * 4;
      const int j0 = (mt >> 6) * 4;
      const float4 r0 = *(const float4*)(kg + (size_t)(jb + j0 + 0) * D1_ + a0);
      const float4 r1 = *(const float4*)(kg + (size_t)(jb + j0 + 1) * D1_ + a0);
      const float4 r2 = *(const float4*)(kg + (size_t)(jb + j0 + 2) * D1_ + a0);
      const float4 r3 = *(const float4*)(kg + (size_t)(jb + j0 + 3) * D1_ + a0);
      v4bf c;
      c[0] = f2bf(r0.x); c[1] = f2bf(r1.x); c[2] = f2bf(r2.x); c[3] = f2bf(r3.x);
      *(v4bf*)(sKT + (a0 + 0) * BN + j0) = c;
      c[0] = f2bf(r0.y); c[1] = f2bf(r1.y); c[2] = f2bf(r2.y); c[3] = f2bf(r3.y);
      *(v4bf*)(sKT + (a0 + 1) * BN + j0) = c;
      c[0] = f2bf(r0.z); c[1] = f2bf(r1.z); c[2] = f2bf(r2.z); c[3] = f2bf(r3.z);
      *(v4bf*)(sKT + (a0 + 2) * BN + j0) = c;
      c[0] = f2bf(r0.w); c[1] = f2bf(r1.w); c[2] = f2bf(r2.w); c[3] = f2bf(r3.w);
      *(v4bf*)(sKT + (a0 + 3) * BN + j0) = c;
    }
    // ---- stage v (bf16)
    for (int e = tid * 4; e < BN * D2_; e += 1024) {
      const float4 vv = *(const float4*)(vg + (size_t)jb * D2_ + e);
      v4bf w;
      w[0] = f2bf(vv.x); w[1] = f2bf(vv.y); w[2] = f2bf(vv.z); w[3] = f2bf(vv.w);
      *(v4bf*)(sV + e) = w;
    }
    __syncthreads();

    // ---- GEMM1: S(64x64) = qn @ k^T ; wave owns S tiles (rt, 2ch) and (rt, 2ch+1)
    v8f s0 = (v8f){0.f,0.f,0.f,0.f,0.f,0.f,0.f,0.f};
    v8f s1 = (v8f){0.f,0.f,0.f,0.f,0.f,0.f,0.f,0.f};
#pragma unroll
    for (int kc = 0; kc < D1_ / 32; ++kc) {
      const v16bf aF = frag_a(sQ, rt * 16, D1_, kc * 32, lane);
      const v16bf b0 = frag_b(sKT, kc * 32, BN, (2 * ch) * 16, lane);
      const v16bf b1 = frag_b(sKT, kc * 32, BN, (2 * ch + 1) * 16, lane);
      s0 = WMMA_BF16(aF, b0, s0);
      s1 = WMMA_BF16(aF, b1, s1);
    }

    // ---- mask tile has been landing in LDS during GEMM1; wait + make visible
    wait_async0();
    __syncthreads();

    // ---- apply decay mask from LDS, accumulate denominator, round S to bf16
#pragma unroll
    for (int r = 0; r < 8; ++r) {
      const int M = mrow0 + r;
      {
        const int N = (2 * ch) * 16 + (lane & 15);
        const float sv = s0[r] * sM[M * BN + N];
        dsum[r] += sv;
        sS[M * BN + N] = f2bf(sv);
      }
      {
        const int N = (2 * ch + 1) * 16 + (lane & 15);
        const float sv = s1[r] * sM[M * BN + N];
        dsum[r] += sv;
        sS[M * BN + N] = f2bf(sv);
      }
    }
    __syncthreads();

    // ---- GEMM2: numerator(64x512) += S @ v ; wave owns col tiles [16ch, 16ch+16)
#pragma unroll
    for (int kc = 0; kc < 2; ++kc) {
      const v16bf aF = frag_a(sS, rt * 16, BN, kc * 32, lane);
#pragma unroll
      for (int ct = 0; ct < 16; ++ct) {
        const v16bf bF = frag_b(sV, kc * 32, D2_, (ch * 16 + ct) * 16, lane);
        accN[ct] = WMMA_BF16(aF, bF, accN[ct]);
      }
    }
  }

  // ---- denominator reduction across lanes/waves
  __syncthreads();
  if (tid < BM) sRed[tid] = 0.0f;
  __syncthreads();
#pragma unroll
  for (int r = 0; r < 8; ++r) atomicAdd(&sRed[mrow0 + r], dsum[r]);  // ds_add_f32
  __syncthreads();
  float invden[8];
#pragma unroll
  for (int r = 0; r < 8; ++r) invden[r] = 1.0f / fmaxf(fabsf(sRed[mrow0 + r]), 1.0f);
  __syncthreads();
  if (tid < BM) sRed[tid] = 0.0f;
  __syncthreads();

  // ---- o = num/den ; accumulate sum of squares for RMSNorm
  float ss[8] = {0.f,0.f,0.f,0.f,0.f,0.f,0.f,0.f};
#pragma unroll
  for (int ct = 0; ct < 16; ++ct) {
#pragma unroll
    for (int r = 0; r < 8; ++r) {
      const float o = accN[ct][r] * invden[r];
      accN[ct][r] = o;
      ss[r] += o * o;
    }
  }
#pragma unroll
  for (int r = 0; r < 8; ++r) atomicAdd(&sRed[mrow0 + r], ss[r]);
  __syncthreads();

  // ---- RMS scale + transposed store: out[b, c, h, :]
#pragma unroll
  for (int r = 0; r < 8; ++r) {
    const int M = mrow0 + r;
    const float sc = rsqrtf(sRed[M] * (1.0f / (float)D2_) + 1e-6f);
    float* orow = out + ((size_t)(b * C_ + i0 + M) * H_ + h) * D2_ + (lane & 15);
#pragma unroll
    for (int ct = 0; ct < 16; ++ct) {
      orow[(ch * 16 + ct) * 16] = accN[ct][r] * sc;
    }
  }
}

// ---------------------------------------------------------------------------
// Kernel 2: last_unnormlized_kv[b,h] = (k .* w)^T @ v   (256 x 512, K = 2048)
// grid: (4 a-tiles * 4 c-tiles, B*H), block: 256 threads
// ---------------------------------------------------------------------------
__global__ __launch_bounds__(256)
void retention_kv(const float* __restrict__ k, const float* __restrict__ v,
                  const float* __restrict__ mask, float* __restrict__ out_kv) {
  __shared__ bf16_t sA[64 * 64];     //  8 KB   (k.*w)^T chunk: [a][j]
  __shared__ bf16_t sB[64 * 128];    // 16 KB   v chunk: [j][c]

  const int tid  = threadIdx.x;
  const int lane = tid & 31;
  const int wave = tid >> 5;
  const int rt   = wave >> 1;
  const int ch   = wave & 1;
  const int bh   = blockIdx.y;
  const int h    = bh & 7;
  const int a0b  = (blockIdx.x >> 2) * 64;
  const int c0   = (blockIdx.x & 3) * 128;

  const float* kg   = k + (size_t)bh * C_ * D1_;
  const float* vg   = v + (size_t)bh * C_ * D2_;
  const float* wrow = mask + ((size_t)h * C_ + (C_ - 1)) * C_;

  v8f acc[4];
#pragma unroll
  for (int t = 0; t < 4; ++t) acc[t] = (v8f){0.f,0.f,0.f,0.f,0.f,0.f,0.f,0.f};

  for (int jc = 0; jc < C_; jc += 64) {
    __syncthreads();
    // stage (k .* w)^T as bf16 in 4x4 micro-tiles (one per thread)
    {
      const int a0 = (tid & 15) * 4;
      const int j0 = (tid >> 4) * 4;
      const float w0 = wrow[jc + j0 + 0];
      const float w1 = wrow[jc + j0 + 1];
      const float w2 = wrow[jc + j0 + 2];
      const float w3 = wrow[jc + j0 + 3];
      const float4 r0 = *(const float4*)(kg + (size_t)(jc + j0 + 0) * D1_ + a0b + a0);
      const float4 r1 = *(const float4*)(kg + (size_t)(jc + j0 + 1) * D1_ + a0b + a0);
      const float4 r2 = *(const float4*)(kg + (size_t)(jc + j0 + 2) * D1_ + a0b + a0);
      const float4 r3 = *(const float4*)(kg + (size_t)(jc + j0 + 3) * D1_ + a0b + a0);
      v4bf c;
      c[0] = f2bf(r0.x * w0); c[1] = f2bf(r1.x * w1); c[2] = f2bf(r2.x * w2); c[3] = f2bf(r3.x * w3);
      *(v4bf*)(sA + (a0 + 0) * 64 + j0) = c;
      c[0] = f2bf(r0.y * w0); c[1] = f2bf(r1.y * w1); c[2] = f2bf(r2.y * w2); c[3] = f2bf(r3.y * w3);
      *(v4bf*)(sA + (a0 + 1) * 64 + j0) = c;
      c[0] = f2bf(r0.z * w0); c[1] = f2bf(r1.z * w1); c[2] = f2bf(r2.z * w2); c[3] = f2bf(r3.z * w3);
      *(v4bf*)(sA + (a0 + 2) * 64 + j0) = c;
      c[0] = f2bf(r0.w * w0); c[1] = f2bf(r1.w * w1); c[2] = f2bf(r2.w * w2); c[3] = f2bf(r3.w * w3);
      *(v4bf*)(sA + (a0 + 3) * 64 + j0) = c;
    }
    // stage v chunk
    for (int e = tid * 4; e < 64 * 128; e += 1024) {
      const int j = e >> 7;
      const int c = e & 127;
      const float4 vv = *(const float4*)(vg + (size_t)(jc + j) * D2_ + c0 + c);
      v4bf w4;
      w4[0] = f2bf(vv.x); w4[1] = f2bf(vv.y); w4[2] = f2bf(vv.z); w4[3] = f2bf(vv.w);
      *(v4bf*)(sB + j * 128 + c) = w4;
    }
    __syncthreads();

#pragma unroll
    for (int kc = 0; kc < 2; ++kc) {
      const v16bf aF = frag_a(sA, rt * 16, 64, kc * 32, lane);
#pragma unroll
      for (int t = 0; t < 4; ++t) {
        const v16bf bF = frag_b(sB, kc * 32, 128, (ch * 4 + t) * 16, lane);
        acc[t] = WMMA_BF16(aF, bF, acc[t]);
      }
    }
  }

  const int mrow0 = rt * 16 + ((lane >> 4) << 3);
#pragma unroll
  for (int r = 0; r < 8; ++r) {
    float* o = out_kv + ((size_t)bh * D1_ + a0b + mrow0 + r) * D2_ + c0 + (lane & 15);
#pragma unroll
    for (int t = 0; t < 4; ++t) o[(ch * 4 + t) * 16] = acc[t][r];
  }
}

// ---------------------------------------------------------------------------
// Kernel 3: last_unnormlized_gk[b,h,a] = sum_j k[b,h,j,a] * w[h,j]
// ---------------------------------------------------------------------------
__global__ __launch_bounds__(256)
void retention_gk(const float* __restrict__ k, const float* __restrict__ mask,
                  float* __restrict__ out_gk) {
  const int bh = blockIdx.x;
  const int h  = bh & 7;
  const int a  = threadIdx.x;                 // 256 threads == D1
  const float* kg   = k + (size_t)bh * C_ * D1_ + a;
  const float* wrow = mask + ((size_t)h * C_ + (C_ - 1)) * C_;
  float s = 0.0f;
  for (int j = 0; j < C_; ++j) s += kg[(size_t)j * D1_] * wrow[j];
  out_gk[(size_t)bh * D1_ + a] = s;
}

// ---------------------------------------------------------------------------
extern "C" void kernel_launch(void* const* d_in, const int* in_sizes, int n_in,
                              void* d_out, int out_size, void* d_ws, size_t ws_size,
                              hipStream_t stream) {
  (void)in_sizes; (void)n_in; (void)out_size; (void)d_ws; (void)ws_size;
  const float* q     = (const float*)d_in[0];
  const float* k     = (const float*)d_in[1];
  const float* v     = (const float*)d_in[2];
  // d_in[3] = chunk_gamma (unused in this forward path)
  const float* mask  = (const float*)d_in[4];
  const float* norml = (const float*)d_in[5];

  float* out    = (float*)d_out;
  float* out_o  = out;
  float* out_kv = out + (size_t)B_ * C_ * H_ * D2_;                 // 33,554,432
  float* out_gk = out_kv + (size_t)B_ * H_ * D1_ * D2_;             // +4,194,304

  dim3 g1(C_ / BM, B_ * H_);     // (32, 32)
  retention_main<<<g1, 256, 0, stream>>>(q, k, v, mask, norml, out_o);

  dim3 g2((D1_ / 64) * (D2_ / 128), B_ * H_);   // (16, 32)
  retention_kv<<<g2, 256, 0, stream>>>(k, v, mask, out_kv);

  retention_gk<<<B_ * H_, D1_, 0, stream>>>(k, mask, out_gk);
}
